// LSTMRegressor_60790967108348
// MI455X (gfx1250) — matrix operand order repaired and checked by reading, not compiled, not run on hardware
//
#include <hip/hip_runtime.h>
#include <stdint.h>

typedef __bf16 bf16;
typedef __attribute__((ext_vector_type(8)))  float v8f;
typedef __attribute__((ext_vector_type(16))) bf16  v16bf;
typedef __attribute__((ext_vector_type(8)))  bf16  v8bf;

#define B_   256
#define T_   512
#define D_   64
#define H_   1024
#define F_   8
#define Q_   3
#define KTOT (D_ + H_)   // 1088
#define NKST (KTOT / 32) // 34 K-steps

// ---------- CDNA5 async global->LDS staging via inline asm (ASYNCcnt path) ----------
// VDST = per-lane LDS byte address (flat shared addr low 32 bits, ISA 10.2),
// VADDR = 64-bit global address, SADDR = off (GV mode).
__device__ __forceinline__ void cp16_g2l(bf16* l, const bf16* g) {
#if defined(__gfx1250__)
    uint32_t lds_addr = (uint32_t)(uintptr_t)l;
    asm volatile("global_load_async_to_lds_b128 %0, %1, off"
                 :
                 : "v"(lds_addr), "v"(g)
                 : "memory");
#else
    *(v8bf*)l = *(const v8bf*)g;
#endif
}

__device__ __forceinline__ void async_stage_wait() {
#if defined(__gfx1250__)
    asm volatile("s_wait_asynccnt 0x0" ::: "memory");
#endif
}

// ---------------- one-time pack kernels ----------------

// Fused transposed weight: Wt[n][k] = (k < D ? Wx[k][n] : Wh[k-D][n]), bf16.
__global__ __launch_bounds__(256) void pack_w_kernel(const float* __restrict__ Wx,
                                                     const float* __restrict__ Wh,
                                                     bf16* __restrict__ Wt) {
    int idx = blockIdx.x * 256 + threadIdx.x;          // 4096*1088 total, exact grid
    int n = idx / KTOT;
    int k = idx % KTOT;
    float v = (k < D_) ? Wx[(size_t)k * (4 * H_) + n]
                       : Wh[(size_t)(k - D_) * (4 * H_) + n];
    Wt[idx] = (bf16)v;
}

__global__ __launch_bounds__(256) void pack_x_kernel(const float* __restrict__ x,
                                                     bf16* __restrict__ xbf) {
    int idx = blockIdx.x * 256 + threadIdx.x;          // B*T*D total, exact grid
    xbf[idx] = (bf16)x[idx];
}

__global__ __launch_bounds__(256) void init_state_kernel(bf16* __restrict__ h0,
                                                         float* __restrict__ c) {
    int idx = blockIdx.x * 256 + threadIdx.x;          // B*H total, exact grid
    h0[idx] = (bf16)0.0f;
    c[idx] = 0.0f;
}

// ---------------- per-timestep fused LSTM cell ----------------

__device__ __forceinline__ float sigmoid_f(float v) {
    return 1.0f / (1.0f + __expf(-v));
}

// Stage one 64x32 A tile + 4x (32x32) B tiles into the given LDS buffers.
__device__ __forceinline__ void stage_tile(int k0, int t, int m0, int n0,
                                           int ar, int ak, int tid,
                                           const bf16* __restrict__ xbf,
                                           const bf16* __restrict__ hin,
                                           const bf16* __restrict__ Wt,
                                           bf16* __restrict__ A,
                                           bf16* __restrict__ Bt) {
    const bf16* asrc =
        (k0 < D_)
            ? xbf + (size_t)(m0 + ar) * (T_ * D_) + (size_t)t * D_ + k0 + ak
            : hin + (size_t)(m0 + ar) * H_ + (k0 - D_) + ak;
    cp16_g2l(&A[ar * 32 + ak], asrc);
#pragma unroll
    for (int cc = 0; cc < 2; ++cc) {
        int c  = tid * 2 + cc;         // 0..511
        int g  = c >> 7;               // gate
        int r  = (c & 127) >> 2;       // n within tile, 0..31
        int kk = (c & 3) * 8;          // 0,8,16,24
        cp16_g2l(&Bt[g * (32 * 32) + r * 32 + kk],
                 Wt + (size_t)(g * H_ + n0 + r) * KTOT + k0 + kk);
    }
}

// Grid (4, 32): blockIdx.x -> 64 batch rows, blockIdx.y -> 32 hidden columns.
// Each block computes z-tiles for all 4 gates and applies the cell update.
__global__ __launch_bounds__(256)
void lstm_step_kernel(const bf16* __restrict__ xbf,   // (B, T, D) bf16
                      const bf16* __restrict__ Wt,    // (4H, KTOT) bf16 pre-transposed
                      const float* __restrict__ bias, // (4H,) fp32
                      const bf16* __restrict__ hin,   // (B, H) bf16
                      bf16* __restrict__ hout,        // (B, H) bf16
                      float* __restrict__ cst,        // (B, H) fp32, in place
                      int t) {
    // double-buffered tiles: 2 * (4KB A + 8KB B) = 24 KB
    __shared__ __attribute__((aligned(16))) bf16 ldsA[2][64 * 32];
    __shared__ __attribute__((aligned(16))) bf16 ldsB[2][4 * 32 * 32];

    const int tid  = threadIdx.x;
    const int lane = tid & 31;
    const int w    = tid >> 5;   // wave id 0..7
    const int wm   = w & 3;      // M sub-tile (16 rows each)
    const int wn   = w >> 2;     // N sub-tile (16 cols each)
    const int m0   = blockIdx.x * 64;
    const int n0   = blockIdx.y * 32;

    v8f acc[4];
#pragma unroll
    for (int g = 0; g < 4; ++g) acc[g] = (v8f){0.f, 0.f, 0.f, 0.f, 0.f, 0.f, 0.f, 0.f};

    const int ar = tid >> 2;          // 0..63 (A tile row)
    const int ak = (tid & 3) * 8;     // 0,8,16,24

    const int arow = wm * 16 + (lane & 15);
    const int kbA  = (lane < 16) ? 0 : 8;
    const int brow = wn * 16 + (lane & 15);
    const int kbB  = (lane < 16) ? 0 : 8;

    // prologue: stage tile 0 into buffer 0
    stage_tile(0, t, m0, n0, ar, ak, tid, xbf, hin, Wt, ldsA[0], ldsB[0]);
    async_stage_wait();
    __syncthreads();

    for (int ks = 0; ks < NKST; ++ks) {
        const int cur = ks & 1;

        // stage next tile into the other buffer while computing this one
        if (ks + 1 < NKST) {
            stage_tile((ks + 1) * 32, t, m0, n0, ar, ak, tid, xbf, hin, Wt,
                       ldsA[cur ^ 1], ldsB[cur ^ 1]);
        }

        // ---- A fragment (ISA 16-bit A 16x32 layout) ----
        const bf16* Ab = ldsA[cur];
        v8bf alo = *(const v8bf*)&Ab[arow * 32 + kbA];
        v8bf ahi = *(const v8bf*)&Ab[arow * 32 + kbA + 16];
        v16bf afrag = __builtin_shufflevector(alo, ahi,
            0, 1, 2, 3, 4, 5, 6, 7, 8, 9, 10, 11, 12, 13, 14, 15);

        // ---- 4 WMMAs: one per gate ----
        const bf16* Bb = ldsB[cur];
#pragma unroll
        for (int g = 0; g < 4; ++g) {
            v8bf blo = *(const v8bf*)&Bb[g * (32 * 32) + brow * 32 + kbB];
            v8bf bhi = *(const v8bf*)&Bb[g * (32 * 32) + brow * 32 + kbB + 16];
            v16bf bfrag = __builtin_shufflevector(blo, bhi,
                0, 1, 2, 3, 4, 5, 6, 7, 8, 9, 10, 11, 12, 13, 14, 15);
            acc[g] = __builtin_amdgcn_wmma_f32_16x16x32_bf16(
                false, afrag, false, bfrag, (short)0, acc[g], false, false);
        }

        // next buffer fully landed + everyone done reading current buffer
        async_stage_wait();
        __syncthreads();
    }

    // ---- epilogue: bias + LSTM cell update (fp32) ----
    const int ncol  = n0 + wn * 16 + (lane & 15);       // hidden column
    const int mbase = m0 + wm * 16 + ((lane < 16) ? 0 : 8);
    const float b_i = bias[0 * H_ + ncol];
    const float b_f = bias[1 * H_ + ncol];
    const float b_g = bias[2 * H_ + ncol];
    const float b_o = bias[3 * H_ + ncol];

#pragma unroll
    for (int r = 0; r < 8; ++r) {
        const int m = mbase + r;
        const size_t idx = (size_t)m * H_ + ncol;
        float zi = acc[0][r] + b_i;
        float zf = acc[1][r] + b_f;
        float zg = acc[2][r] + b_g;
        float zo = acc[3][r] + b_o;
        float ig = sigmoid_f(zi);
        float fg = sigmoid_f(zf);
        float gg = tanhf(zg);
        float og = sigmoid_f(zo);
        float cn = fg * cst[idx] + ig * gg;
        cst[idx]  = cn;
        hout[idx] = (bf16)(og * tanhf(cn));
    }
}

// ---------------- tiny per-feature regression head ----------------
__global__ __launch_bounds__(256)
void head_kernel(const bf16* __restrict__ h, const float* __restrict__ Wr,
                 const float* __restrict__ br, float* __restrict__ out) {
    int idx = blockIdx.x * 256 + threadIdx.x;   // 256*24 = 6144, exact grid
    int b  = idx / (F_ * Q_);
    int fq = idx % (F_ * Q_);
    int f  = fq / Q_;
    int q  = fq % Q_;
    float sum = br[fq];
    const bf16* hb = h + (size_t)b * H_;
    const float* wr = Wr + (size_t)f * H_ * Q_ + q;
#pragma unroll 4
    for (int k = 0; k < H_; ++k) {
        sum += (float)hb[k] * wr[(size_t)k * Q_];
    }
    out[idx] = sum;
}

// ---------------- host-side launch ----------------

extern "C" void kernel_launch(void* const* d_in, const int* in_sizes, int n_in,
                              void* d_out, int out_size, void* d_ws, size_t ws_size,
                              hipStream_t stream) {
    (void)in_sizes; (void)n_in; (void)out_size; (void)ws_size;

    const float* x    = (const float*)d_in[0];   // (B,T,D)
    const float* Wx   = (const float*)d_in[1];   // (D,4H)
    const float* Wh   = (const float*)d_in[2];   // (H,4H)
    const float* bias = (const float*)d_in[3];   // (4H,)
    const float* Wr   = (const float*)d_in[4];   // (F,H,Q)
    const float* br   = (const float*)d_in[5];   // (F,Q)
    float* out = (float*)d_out;

    char* ws = (char*)d_ws;
    // workspace layout (all 256B aligned), ~27.8 MB total
    bf16*  Wt  = (bf16*)(ws + 0);          // 4096*1088*2 = 8,912,896
    bf16*  xbf = (bf16*)(ws + 8912896);    // 256*512*64*2 = 16,777,216
    bf16*  h0  = (bf16*)(ws + 25690112);   // 256*1024*2 = 524,288
    bf16*  h1  = (bf16*)(ws + 26214400);   // 524,288
    float* c   = (float*)(ws + 26738688);  // 256*1024*4 = 1,048,576

    // one-time packs / init (re-done every call: no cross-call state)
    pack_w_kernel<<<(4 * H_ * KTOT) / 256, 256, 0, stream>>>(Wx, Wh, Wt);
    pack_x_kernel<<<(B_ * T_ * D_) / 256, 256, 0, stream>>>(x, xbf);
    init_state_kernel<<<(B_ * H_) / 256, 256, 0, stream>>>(h0, c);

    // sequential recurrence: one fused GEMM+cell kernel per timestep, ping-pong h
    dim3 grid(B_ / 64, H_ / 32);   // (4, 32) = 128 blocks
    for (int t = 0; t < T_; ++t) {
        const bf16* hin = (t & 1) ? h1 : h0;
        bf16* hout      = (t & 1) ? h0 : h1;
        lstm_step_kernel<<<grid, 256, 0, stream>>>(xbf, Wt, bias, hin, hout, c, t);
    }
    // T=512 even -> final hidden lands in h0
    head_kernel<<<(B_ * F_ * Q_) / 256, 256, 0, stream>>>(h0, Wr, br, out);
}
